// MultiHeadAttentionLayer_53498112639054
// MI455X (gfx1250) — matrix operand order repaired and checked by reading, not compiled, not run on hardware
//
#include <hip/hip_runtime.h>
#include <hip/hip_bf16.h>

// ---------------------------------------------------------------------------
// Types for CDNA5 WMMA (wave32): A/B are 16 bf16 elems/lane, C/D 8 f32/lane.
// ---------------------------------------------------------------------------
typedef __attribute__((ext_vector_type(16))) __bf16 v16bf;
typedef __attribute__((ext_vector_type(8)))  float  v8f;
typedef __attribute__((ext_vector_type(16))) short  v16s;
typedef __attribute__((ext_vector_type(8)))  short  v8s;

union FragAB {
    struct { v8s lo, hi; } h;
    v16s s;
    v16bf b;
};

__device__ __forceinline__ unsigned short f2bf(float x) {
    unsigned u = __float_as_uint(x);
    u += 0x7FFFu + ((u >> 16) & 1u);   // round-to-nearest-even
    return (unsigned short)(u >> 16);
}

__device__ __forceinline__ v8f wmma_bf16(const FragAB& a, const FragAB& b, v8f c) {
    return __builtin_amdgcn_wmma_f32_16x16x32_bf16(
        /*neg_a=*/false, a.b, /*neg_b=*/false, b.b,
        /*c_mod=*/(short)0, c, /*reuse_a=*/false, /*reuse_b=*/false);
}

// ---------------------------------------------------------------------------
// CDNA5 async copy: global -> LDS, 16B per lane, tracked by ASYNCcnt.
// Toolchain declares the builtin with (AS1 int-vec16*, AS3 int-vec16*, imm,
// imm) — match that exactly (diagnostic from round 2 showed the param type).
// ---------------------------------------------------------------------------
typedef int i128v __attribute__((vector_size(16)));

__device__ __forceinline__ void async_b128(const void* g, void* l) {
#if __has_builtin(__builtin_amdgcn_global_load_async_to_lds_b128)
    __builtin_amdgcn_global_load_async_to_lds_b128(
        (__attribute__((address_space(1))) i128v*)(void*)g,
        (__attribute__((address_space(3))) i128v*)l,
        /*imm offset=*/0, /*cpol=*/0);
#else
    unsigned loff = (unsigned)(unsigned long long)
        (__attribute__((address_space(3))) void*)l;
    asm volatile("global_load_async_to_lds_b128 %0, %1, off"
                 :: "v"(loff), "v"(g) : "memory");
#endif
}

__device__ __forceinline__ void wait_async0() {
#if __has_builtin(__builtin_amdgcn_s_wait_asynccnt)
    __builtin_amdgcn_s_wait_asynccnt(0);
#else
    asm volatile("s_wait_asynccnt 0x0" ::: "memory");
#endif
}

// ---------------------------------------------------------------------------
// Problem constants
// ---------------------------------------------------------------------------
#define BATCH 8
#define SEQ   1024
#define HID   1024
#define HEADS 16
#define HDIM  64
#define TOK   (BATCH * SEQ)      // 8192

// ---------------------------------------------------------------------------
// Kernel 0: f32 -> bf16 convert
// ---------------------------------------------------------------------------
__global__ void k_f32_to_bf16(const float* __restrict__ src,
                              unsigned short* __restrict__ dst, int n) {
    int i = blockIdx.x * blockDim.x + threadIdx.x;
    if (i < n) dst[i] = f2bf(src[i]);
}

// ---------------------------------------------------------------------------
// Kernel 1: GEMM  C[M,N] = A[M,K] * Bw[N,K]^T + bias   (bf16 in, f32 out)
// Block = 256 threads (8 waves); block tile 64(M) x 128(N), K step 32.
// Double-buffered LDS staging via global_load_async_to_lds_b128 (ASYNCcnt):
//   stage = A tile 64x32 (4KB) + B tile 128x32 (8KB) = 12KB = 3 b128/thread.
// Waves: (wm = wave&3) -> 16-row strip, (wn = wave>>2) -> 64-col strip,
// 4 C-tiles / wave, fragments read from LDS (ds_load_b128).
// Fragment layouts (A and B use one contiguous k order):
//   A (16x32): lane row = lane%16; k = koff + e (16 contiguous),
//              koff = 16*(lane>=16)
//   B (32x16): lane col = lane%16; k = 16*(lane/16) + e (16 contiguous)
//   C:         lane n = lane%16, VGPR v -> m = v + 8*(lane/16)
// ---------------------------------------------------------------------------
#define GBM 64
#define GBN 128
#define GBK 32

__global__ __launch_bounds__(256) void k_gemm_bf16(
    const unsigned short* __restrict__ A,   // [M x K] bf16
    const unsigned short* __restrict__ Bw,  // [N x K] bf16
    const float* __restrict__ bias,         // [N]
    float* __restrict__ C,                  // [M x N] f32
    int M, int N, int K) {
    __shared__ unsigned short ldsA[2][GBM][GBK];   // 2 x 4KB
    __shared__ unsigned short ldsB[2][GBN][GBK];   // 2 x 8KB

    const int tid  = threadIdx.x;
    const int lane = tid & 31;
    const int wave = tid >> 5;
    const int wm   = wave & 3;
    const int wn   = wave >> 2;
    const int rowBase = blockIdx.x * GBM;
    const int colBase = blockIdx.y * GBN;
    const int ln15 = lane & 15;
    const int half = lane >> 4;
    const int koffA = half * 16;
    const int koffB = half * 16;

    // async-copy mapping: A = 256 16B-chunks (1/thread), B = 512 (2/thread)
    const int ar = tid >> 2;            // A row 0..63
    const int ac = (tid & 3) * 8;       // A k-chunk (elements)
    const int br = tid >> 2;            // B rows br and br+64
    const int bc = (tid & 3) * 8;

    v8f acc[4];
#pragma unroll
    for (int t = 0; t < 4; ++t) acc[t] = (v8f){0.f,0.f,0.f,0.f,0.f,0.f,0.f,0.f};

    const int nst = K / GBK;

    // prologue: stage 0
    {
        async_b128(A  + (size_t)(rowBase + ar) * K + ac,        &ldsA[0][ar][ac]);
        async_b128(Bw + (size_t)(colBase + br) * K + bc,        &ldsB[0][br][bc]);
        async_b128(Bw + (size_t)(colBase + 64 + br) * K + bc,   &ldsB[0][64 + br][bc]);
    }

    for (int i = 0; i < nst; ++i) {
        const int st = i & 1;
        wait_async0();          // our stage-i chunks landed in LDS
        __syncthreads();        // everyone's chunks landed; prev reads done

        if (i + 1 < nst) {      // overlap stage i+1 DMA with stage i math
            const int ns = st ^ 1;
            const int kb = (i + 1) * GBK;
            async_b128(A  + (size_t)(rowBase + ar) * K + kb + ac,      &ldsA[ns][ar][ac]);
            async_b128(Bw + (size_t)(colBase + br) * K + kb + bc,      &ldsB[ns][br][bc]);
            async_b128(Bw + (size_t)(colBase + 64 + br) * K + kb + bc, &ldsB[ns][64 + br][bc]);
        }

        FragAB af;
        af.h.lo = *reinterpret_cast<const v8s*>(&ldsA[st][wm * 16 + ln15][koffA]);
        af.h.hi = *reinterpret_cast<const v8s*>(&ldsA[st][wm * 16 + ln15][koffA + 8]);
#pragma unroll
        for (int t = 0; t < 4; ++t) {
            FragAB bf_;
            bf_.h.lo = *reinterpret_cast<const v8s*>(&ldsB[st][wn * 64 + t * 16 + ln15][koffB]);
            bf_.h.hi = *reinterpret_cast<const v8s*>(&ldsB[st][wn * 64 + t * 16 + ln15][koffB + 8]);
            acc[t] = wmma_bf16(af, bf_, acc[t]);
        }
    }

#pragma unroll
    for (int t = 0; t < 4; ++t) {
        const int col = colBase + t * 16 + ln15;
        const float bv = bias[col];
#pragma unroll
        for (int v = 0; v < 8; ++v) {
            const int row = rowBase + v + 8 * half;
            C[(size_t)row * N + col] = acc[t][v] + bv;
        }
    }
}

// ---------------------------------------------------------------------------
// Kernel 2: RoPE (reference semantics: pos_emb = [sin(a), cos(a)], then
// cos/sin applied AGAIN to the table values; applied to q, k AND v).
// Writes q,k bf16 [B,H,S,D]; v bf16 TRANSPOSED [B,H,D,S] so the attention
// P*V B-fragment (k = key index contiguous per lane) is two b128 loads.
// ---------------------------------------------------------------------------
__global__ void k_rope(const float* __restrict__ pq,
                       const float* __restrict__ pk,
                       const float* __restrict__ pv,
                       unsigned short* __restrict__ qb,
                       unsigned short* __restrict__ kb,
                       unsigned short* __restrict__ vt) {
    int idx = blockIdx.x * blockDim.x + threadIdx.x;   // B*S*512
    if (idx >= BATCH * SEQ * (HID / 2)) return;
    const int i = idx & 511;          // freq / low-channel index
    const int s = (idx >> 9) & 1023;  // seq position
    const int b = idx >> 19;          // batch

    const float invf = __powf(10000.f, -(float)i * (1.f / 512.f));
    const float a    = (float)s * invf;
    const float sv   = __sinf(a);     // p1 table value
    const float cv   = __cosf(a);     // p2 table value
    const float c1 = __cosf(sv), s1 = __sinf(sv);
    const float c2 = __cosf(cv), s2 = __sinf(cv);

    const size_t base = ((size_t)b * SEQ + s) * HID;
    const int h1 = i >> 6, d = i & 63, h2 = h1 + 8;
    const size_t rowLo = (((size_t)b * HEADS + h1) * SEQ + s) * HDIM + d;
    const size_t rowHi = (((size_t)b * HEADS + h2) * SEQ + s) * HDIM + d;
    const size_t colLo = (((size_t)b * HEADS + h1) * HDIM + d) * SEQ + s;
    const size_t colHi = (((size_t)b * HEADS + h2) * HDIM + d) * SEQ + s;

    {   // Q
        const float x1 = pq[base + i], x2 = pq[base + i + 512];
        qb[rowLo] = f2bf(x1 * c1 + x2 * s1);
        qb[rowHi] = f2bf(x2 * c2 - x1 * s2);
    }
    {   // K
        const float x1 = pk[base + i], x2 = pk[base + i + 512];
        kb[rowLo] = f2bf(x1 * c1 + x2 * s1);
        kb[rowHi] = f2bf(x2 * c2 - x1 * s2);
    }
    {   // V (transposed store)
        const float x1 = pv[base + i], x2 = pv[base + i + 512];
        vt[colLo] = f2bf(x1 * c1 + x2 * s1);
        vt[colHi] = f2bf(x2 * c2 - x1 * s2);
    }
}

// ---------------------------------------------------------------------------
// Kernel 3: flash attention. Block = 128 threads (4 waves); each wave owns a
// 16-row q tile; block covers 64 q rows of one (b,h). 32-key inner tiles:
// S = Q*K^T (2 wmma, K-dim 64), online softmax in f32 with shfl_xor
// reductions inside 16-lane halves, P re-swizzled C->A layout via a
// PER-WAVE LDS region (LDS pipe is in-order per wave -> no block barrier
// needed), then P*V (4 wmma, N over D=64). Output ctx bf16 in [B,S,HID].
// ---------------------------------------------------------------------------
__global__ __launch_bounds__(128) void k_attn(
    const unsigned short* __restrict__ qbuf,   // [B,H,S,D] bf16
    const unsigned short* __restrict__ kbuf,   // [B,H,S,D] bf16
    const unsigned short* __restrict__ vt,     // [B,H,D,S] bf16
    unsigned short* __restrict__ ctx) {        // [B,S,HID] bf16
    __shared__ float ldsP[4][16][32];

    const int lane = threadIdx.x & 31;
    const int wave = threadIdx.x >> 5;
    const int bid  = blockIdx.x;
    const int qt   = bid & 15;
    const int h    = (bid >> 4) & 15;
    const int b    = bid >> 8;
    const int qrow0 = qt * 64 + wave * 16;
    const int ln15 = lane & 15;
    const int half = lane >> 4;
    const size_t bh = (size_t)b * HEADS + h;

    // Q fragments (K-dim 64 -> two 16x32 A-frags), k = 16*half + e per lane
    const unsigned short* qp = qbuf + (bh * SEQ + qrow0 + ln15) * HDIM;
    FragAB aq0, aq1;
    aq0.h.lo = *reinterpret_cast<const v8s*>(qp + half * 16);
    aq0.h.hi = *reinterpret_cast<const v8s*>(qp + half * 16 + 8);
    aq1.h.lo = *reinterpret_cast<const v8s*>(qp + 32 + half * 16);
    aq1.h.hi = *reinterpret_cast<const v8s*>(qp + 32 + half * 16 + 8);

    float mrow[8], lrow[8];
    v8f acc[4];
#pragma unroll
    for (int v = 0; v < 8; ++v) { mrow[v] = -1e30f; lrow[v] = 0.f; }
#pragma unroll
    for (int t = 0; t < 4; ++t) acc[t] = (v8f){0.f,0.f,0.f,0.f,0.f,0.f,0.f,0.f};

    const float scale = 0.125f;   // 1/sqrt(64)

    for (int j = 0; j < SEQ; j += 32) {
        v8f s1 = (v8f){0.f,0.f,0.f,0.f,0.f,0.f,0.f,0.f};
        v8f s2 = (v8f){0.f,0.f,0.f,0.f,0.f,0.f,0.f,0.f};

        // keys j..j+15 -> s1
        {
            const unsigned short* kp = kbuf + (bh * SEQ + j + ln15) * HDIM;
            FragAB bk;
            bk.h.lo = *reinterpret_cast<const v8s*>(kp + half * 16);
            bk.h.hi = *reinterpret_cast<const v8s*>(kp + half * 16 + 8);
            s1 = wmma_bf16(aq0, bk, s1);
            bk.h.lo = *reinterpret_cast<const v8s*>(kp + 32 + half * 16);
            bk.h.hi = *reinterpret_cast<const v8s*>(kp + 32 + half * 16 + 8);
            s1 = wmma_bf16(aq1, bk, s1);
        }
        // keys j+16..j+31 -> s2
        {
            const unsigned short* kp = kbuf + (bh * SEQ + j + 16 + ln15) * HDIM;
            FragAB bk;
            bk.h.lo = *reinterpret_cast<const v8s*>(kp + half * 16);
            bk.h.hi = *reinterpret_cast<const v8s*>(kp + half * 16 + 8);
            s2 = wmma_bf16(aq0, bk, s2);
            bk.h.lo = *reinterpret_cast<const v8s*>(kp + 32 + half * 16);
            bk.h.hi = *reinterpret_cast<const v8s*>(kp + 32 + half * 16 + 8);
            s2 = wmma_bf16(aq1, bk, s2);
        }

        // online softmax over the 32 columns of this tile, per row
#pragma unroll
        for (int v = 0; v < 8; ++v) {
            float a1 = s1[v] * scale, a2 = s2[v] * scale;
            float mx = fmaxf(a1, a2);
#pragma unroll
            for (int m = 1; m <= 8; m <<= 1) mx = fmaxf(mx, __shfl_xor(mx, m, 32));
            const float mnew = fmaxf(mrow[v], mx);
            const float corr = __expf(mrow[v] - mnew);
            const float p1 = __expf(a1 - mnew);
            const float p2 = __expf(a2 - mnew);
            float rs = p1 + p2;
#pragma unroll
            for (int m = 1; m <= 8; m <<= 1) rs += __shfl_xor(rs, m, 32);
            lrow[v] = lrow[v] * corr + rs;
            mrow[v] = mnew;
#pragma unroll
            for (int t = 0; t < 4; ++t) acc[t][v] *= corr;
            const int mr = v + 8 * half;          // C-layout row
            ldsP[wave][mr][ln15]      = p1;
            ldsP[wave][mr][16 + ln15] = p2;
        }

        // re-swizzle P: C layout -> A layout (16x32 bf16 fragment).
        // Per-wave LDS region + in-order LDS pipe => no barrier needed.
        FragAB ap;
        const float* prow = &ldsP[wave][ln15][0];
#pragma unroll
        for (int e = 0; e < 8; ++e) {
            ap.s[e]     = (short)f2bf(prow[half * 16 + e]);
            ap.s[8 + e] = (short)f2bf(prow[half * 16 + 8 + e]);
        }

        // acc += P * V  (V transposed: [B,H,D,S])
#pragma unroll
        for (int t = 0; t < 4; ++t) {
            const unsigned short* vp =
                vt + ((bh * HDIM + t * 16 + ln15) * SEQ + j + half * 16);
            FragAB bv;
            bv.h.lo = *reinterpret_cast<const v8s*>(vp);
            bv.h.hi = *reinterpret_cast<const v8s*>(vp + 8);
            acc[t] = wmma_bf16(ap, bv, acc[t]);
        }
    }

    // normalize and store context in [B,S,HID] bf16 for the output GEMM
#pragma unroll
    for (int t = 0; t < 4; ++t) {
#pragma unroll
        for (int v = 0; v < 8; ++v) {
            const int mr  = v + 8 * half;
            const int tok = qrow0 + mr;
            const float val = acc[t][v] / lrow[v];
            ctx[((size_t)b * SEQ + tok) * HID + h * HDIM + t * 16 + ln15] = f2bf(val);
        }
    }
}

// ---------------------------------------------------------------------------
// Host-side launch
// ---------------------------------------------------------------------------
extern "C" void kernel_launch(void* const* d_in, const int* in_sizes, int n_in,
                              void* d_out, int out_size, void* d_ws, size_t ws_size,
                              hipStream_t stream) {
    const float* query = (const float*)d_in[0];
    const float* keyi  = (const float*)d_in[1];
    const float* value = (const float*)d_in[2];
    const float* Wq = (const float*)d_in[3];
    const float* bq = (const float*)d_in[4];
    const float* Wk = (const float*)d_in[5];
    const float* bk = (const float*)d_in[6];
    const float* Wv = (const float*)d_in[7];
    const float* bv = (const float*)d_in[8];
    const float* Wo = (const float*)d_in[9];
    const float* bo = (const float*)d_in[10];

    char* ws = (char*)d_ws;
    const size_t ACT_B = (size_t)TOK * HID * sizeof(unsigned short);   // 16 MiB
    const size_t W_B   = (size_t)HID * HID * sizeof(unsigned short);   //  2 MiB
    const size_t PRJ_B = (size_t)TOK * HID * sizeof(float);            // 32 MiB

    unsigned short* actq = (unsigned short*)(ws);
    unsigned short* actk = (unsigned short*)(ws + ACT_B);
    unsigned short* actv = (unsigned short*)(ws + 2 * ACT_B);
    unsigned short* wq   = (unsigned short*)(ws + 3 * ACT_B);
    unsigned short* wk   = (unsigned short*)(ws + 3 * ACT_B + W_B);
    unsigned short* wv   = (unsigned short*)(ws + 3 * ACT_B + 2 * W_B);
    unsigned short* wo   = (unsigned short*)(ws + 3 * ACT_B + 3 * W_B);
    float* projq = (float*)(ws + 3 * ACT_B + 4 * W_B);
    float* projk = (float*)(ws + 3 * ACT_B + 4 * W_B + PRJ_B);
    float* projv = (float*)(ws + 3 * ACT_B + 4 * W_B + 2 * PRJ_B);
    unsigned short* qbuf = (unsigned short*)(ws + 3 * ACT_B + 4 * W_B + 3 * PRJ_B);
    unsigned short* kbuf = (unsigned short*)(ws + 4 * ACT_B + 4 * W_B + 3 * PRJ_B);
    unsigned short* vtbf = (unsigned short*)(ws + 5 * ACT_B + 4 * W_B + 3 * PRJ_B);
    unsigned short* ctx  = actq;   // activations dead by the time attention runs

    const int NA = TOK * HID;      // 8388608
    const int NW = HID * HID;      // 1048576
    k_f32_to_bf16<<<(NA + 255) / 256, 256, 0, stream>>>(query, actq, NA);
    k_f32_to_bf16<<<(NA + 255) / 256, 256, 0, stream>>>(keyi,  actk, NA);
    k_f32_to_bf16<<<(NA + 255) / 256, 256, 0, stream>>>(value, actv, NA);
    k_f32_to_bf16<<<(NW + 255) / 256, 256, 0, stream>>>(Wq, wq, NW);
    k_f32_to_bf16<<<(NW + 255) / 256, 256, 0, stream>>>(Wk, wk, NW);
    k_f32_to_bf16<<<(NW + 255) / 256, 256, 0, stream>>>(Wv, wv, NW);
    k_f32_to_bf16<<<(NW + 255) / 256, 256, 0, stream>>>(Wo, wo, NW);

    dim3 ggrid(TOK / 64, HID / 128);
    k_gemm_bf16<<<ggrid, 256, 0, stream>>>(actq, wq, bq, projq, TOK, HID, HID);
    k_gemm_bf16<<<ggrid, 256, 0, stream>>>(actk, wk, bk, projk, TOK, HID, HID);
    k_gemm_bf16<<<ggrid, 256, 0, stream>>>(actv, wv, bv, projv, TOK, HID, HID);

    const int NR = BATCH * SEQ * (HID / 2);
    k_rope<<<(NR + 255) / 256, 256, 0, stream>>>(projq, projk, projv, qbuf, kbuf, vtbf);

    k_attn<<<BATCH * HEADS * (SEQ / 64), 128, 0, stream>>>(qbuf, kbuf, vtbf, ctx);

    k_gemm_bf16<<<ggrid, 256, 0, stream>>>(ctx, wo, bo, (float*)d_out, TOK, HID, HID);
}
